// CrossFrameAttention_49735721288273
// MI455X (gfx1250) — matrix operand order repaired and checked by reading, not compiled, not run on hardware
//
#include <hip/hip_runtime.h>

// ---------------------------------------------------------------------------
// CDNA5 (gfx1250) fused cross-frame attention.
//   FB=8 batches, C=256 channels, N=4096 tokens, QK=32.
//   Matmuls on v_wmma_f32_16x16x32_bf16 (wave32); K/V panels staged to LDS
//   with the Tensor Data Mover (double-buffered, TENSORcnt-tracked).
// ---------------------------------------------------------------------------

typedef __bf16 bf16_t;
typedef __attribute__((ext_vector_type(8)))  __bf16 v8bf;
typedef __attribute__((ext_vector_type(16))) __bf16 v16bf;
typedef __attribute__((ext_vector_type(8)))  float  v8f;
typedef __attribute__((ext_vector_type(4)))  unsigned int u32x4;
typedef __attribute__((ext_vector_type(4)))  int i32x4;
typedef __attribute__((ext_vector_type(8)))  int i32x8;

union BF16x16 { v16bf v; v8bf h[2]; };

static __device__ __forceinline__ v8f v8f_zero() {
  v8f z;
#pragma unroll
  for (int i = 0; i < 8; ++i) z[i] = 0.0f;
  return z;
}

#define FBATCH 8
#define CDIM   256
#define NTOK   4096
#define QKDIM  32

// ---------------------------------------------------------------------------
// TDM 2D tile load: build D# groups per CDNA5 ISA and
// issue tensor_load_to_lds.  Units are 4-byte elements (data_size code 2).
// ---------------------------------------------------------------------------
static __device__ __forceinline__ void tdm_load_2d(const void* gptr, void* lptr,
                                                   unsigned tile_d0_units,
                                                   unsigned tile_d1,
                                                   unsigned tensor_d0_units,
                                                   unsigned tensor_d1,
                                                   unsigned stride_units) {
  const unsigned long long ga = (unsigned long long)(uintptr_t)gptr;
  const unsigned la = (unsigned)(uintptr_t)lptr;     // LDS byte offset (addr[31:0])
  u32x4 g0;
  g0[0] = 1u;                                        // count=1, user-mode, no gather
  g0[1] = la;                                        // lds_addr
  g0[2] = (unsigned)(ga & 0xFFFFFFFFu);              // global_addr[31:0]
  g0[3] = (unsigned)((ga >> 32) & 0x01FFFFFFu)       // global_addr[56:32]
        | (2u << 30);                                // type = 2 ("image")
  i32x8 g1;
  g1[0] = (int)(2u << 16);                           // wg_mask=0, data_size=2 (4B)
  g1[1] = (int)((tensor_d0_units & 0xFFFFu) << 16);  // atomic_barrier=0 | dim0 lo16
  g1[2] = (int)((tensor_d0_units >> 16) | ((tensor_d1 & 0xFFFFu) << 16));
  g1[3] = (int)((tensor_d1 >> 16) | (tile_d0_units << 16));
  g1[4] = (int)(tile_d1 & 0xFFFFu);                  // tile_dim1 | tile_dim2=0
  g1[5] = (int)stride_units;                         // tensor_dim0_stride[31:0]
  g1[6] = 0;                                         // stride hi | dim1_stride lo
  g1[7] = 0;                                         // dim1_stride hi
  i32x4 z4 = {0, 0, 0, 0};                           // groups 2/3: 2D tensor
#if defined(__clang_major__) && (__clang_major__ >= 23)
  i32x8 z8 = {0, 0, 0, 0, 0, 0, 0, 0};
  __builtin_amdgcn_tensor_load_to_lds(g0, g1, z4, z4, z8, 0);
#else
  __builtin_amdgcn_tensor_load_to_lds(g0, g1, z4, z4, 0);
#endif
}

// ---------------------------------------------------------------------------
// Kernel 0: x [FB][C][N] f32  ->  xT [FB][N][C] bf16  (LDS tiled transpose)
// ---------------------------------------------------------------------------
__global__ void k_transpose_cvt(const float* __restrict__ x, bf16_t* __restrict__ xT) {
  __shared__ float tile[32][33];
  const int b  = blockIdx.z;
  const int n0 = blockIdx.x * 32;
  const int c0 = blockIdx.y * 32;
  const int tx = threadIdx.x, ty = threadIdx.y;      // 32 x 8
  const float* src = x + ((size_t)b * CDIM + c0) * NTOK + n0;
#pragma unroll
  for (int r = 0; r < 32; r += 8)
    tile[ty + r][tx] = src[(size_t)(ty + r) * NTOK + tx];
  __syncthreads();
  bf16_t* dst = xT + ((size_t)b * NTOK + n0) * CDIM + c0;
#pragma unroll
  for (int r = 0; r < 32; r += 8)
    dst[(size_t)(ty + r) * CDIM + tx] = (bf16_t)tile[tx][ty + r];
}

// ---------------------------------------------------------------------------
// Kernel 1: W [R][Ccols] f32 -> WT [Ccols][R] bf16  (tiny weight transposes)
// ---------------------------------------------------------------------------
__global__ void k_transpose_w(const float* __restrict__ W, bf16_t* __restrict__ WT,
                              int R, int Ccols) {
  int i = blockIdx.x * blockDim.x + threadIdx.x;     // output index: [c][r]
  if (i >= R * Ccols) return;
  int r = i % R, c = i / R;
  WT[i] = (bf16_t)W[(size_t)r * Ccols + c];
}

// ---------------------------------------------------------------------------
// Kernel 2: row-major projection.  out[b][n][d] = sum_c xT[b][n][c]*WT[c][d]+bias[d]
//   A-operand (16xK bf16): lane M = lane&15; chunks K = 8*(lane>>4), 16+8*(lane>>4)
//   B-operand (Kx16 bf16): lane N = lane&15; chunks K = 8*(lane>>4), 16+8*(lane>>4)
//   Used for q (Dout=32) and vT (Dout=256). One wave / 16x16 D tile.
// ---------------------------------------------------------------------------
__global__ void k_proj_rowmajor(const bf16_t* __restrict__ xT,
                                const bf16_t* __restrict__ WT,
                                const float* __restrict__ bias,
                                bf16_t* __restrict__ out, int Dout) {
  const int lane = threadIdx.x;
  const int g = lane >> 4, m = lane & 15;
  const int b  = blockIdx.z;
  const int ib = blockIdx.x * 16;                    // token tile
  const int db = blockIdx.y * 16;                    // output-channel tile
  const bf16_t* xrow = xT + ((size_t)b * NTOK + ib + m) * CDIM;
  v8f acc = v8f_zero();
  for (int kc = 0; kc < CDIM; kc += 32) {
    BF16x16 A, B;
    A.h[0] = *(const v8bf*)(xrow + kc + 8 * g);
    A.h[1] = *(const v8bf*)(xrow + kc + 16 + 8 * g);
    const bf16_t* wr = WT + (size_t)kc * Dout + db + m;
#pragma unroll
    for (int e = 0; e < 8; ++e) {
      B.v[e]     = wr[(size_t)(8 * g + e) * Dout];
      B.v[e + 8] = wr[(size_t)(16 + 8 * g + e) * Dout];
    }
    acc = __builtin_amdgcn_wmma_f32_16x16x32_bf16(false, A.v, false, B.v,
                                                  (short)0, acc, false, false);
  }
  const float bb = bias[db + m];
  bf16_t* orow = out + ((size_t)b * NTOK + ib) * (size_t)Dout + db + m;
#pragma unroll
  for (int v = 0; v < 8; ++v)
    orow[(size_t)(v + 8 * g) * Dout] = (bf16_t)(acc[v] + bb);
}

// ---------------------------------------------------------------------------
// Kernel 3: k projection straight into [FB][32][N] (B-operand layout for scores).
// ---------------------------------------------------------------------------
__global__ void k_proj_k(const float* __restrict__ x, const float* __restrict__ Wk,
                         const float* __restrict__ bk, bf16_t* __restrict__ kout) {
  const int lane = threadIdx.x;
  const int g = lane >> 4, m = lane & 15;
  const int b  = blockIdx.z;
  const int nb = blockIdx.x * 16;                    // token tile
  const int db = blockIdx.y * 16;                    // qk-dim tile (0 or 16)
  const float* wrow = Wk + (size_t)(db + m) * CDIM;
  v8f acc = v8f_zero();
  for (int kc = 0; kc < CDIM; kc += 32) {
    BF16x16 A, B;
#pragma unroll
    for (int e = 0; e < 8; ++e) {
      A.v[e]     = (bf16_t)wrow[kc + 8 * g + e];
      A.v[e + 8] = (bf16_t)wrow[kc + 16 + 8 * g + e];
    }
    const float* xrow = x + ((size_t)b * CDIM + kc) * NTOK + nb + m;
#pragma unroll
    for (int e = 0; e < 8; ++e) {
      B.v[e]     = (bf16_t)xrow[(size_t)(8 * g + e) * NTOK];
      B.v[e + 8] = (bf16_t)xrow[(size_t)(16 + 8 * g + e) * NTOK];
    }
    acc = __builtin_amdgcn_wmma_f32_16x16x32_bf16(false, A.v, false, B.v,
                                                  (short)0, acc, false, false);
  }
#pragma unroll
  for (int v = 0; v < 8; ++v) {
    int d = db + v + 8 * g;
    kout[((size_t)b * QKDIM + d) * NTOK + nb + m] = (bf16_t)(acc[v] + bk[d]);
  }
}

// ---------------------------------------------------------------------------
// Kernel 4: fused flash attention + gamma*out + residual.
//   4 waves / workgroup, each owns a 16-query tile (64 rows per WG, so the
//   K tile (2KB) and V panel (16KB) are fetched once per WG per j-step).
//   Wave 0 stages K/V into double-buffered LDS with TENSOR_LOAD_TO_LDS,
//   overlapping DMA of tile t+1 with the 18 WMMAs of tile t; drained with
//   s_wait_tensorcnt + block barrier.
// ---------------------------------------------------------------------------
__global__ void __launch_bounds__(128)
k_flash_attn(const bf16_t* __restrict__ qA, const bf16_t* __restrict__ kM,
             const bf16_t* __restrict__ vT, const float* __restrict__ xin,
             const float* __restrict__ gamma, float* __restrict__ outp) {
  const int tid  = threadIdx.x;
  const int wid  = tid >> 5;
  const int lane = tid & 31;
  const int g = lane >> 4, m = lane & 15;
  const int b  = blockIdx.y;
  const int ib = blockIdx.x * 64 + wid * 16;         // this wave's query tile

  __shared__ bf16_t Kbuf[2][32 * 32];                // [d][j] tile, 2KB each
  __shared__ bf16_t Vbuf[2][32 * 256];               // [j][c] panel, 16KB each
  __shared__ bf16_t Plds[4][16 * 32];                // per-wave P re-layout

  // Q tile in A-operand layout (row = query, K = qk-dim 32).
  BF16x16 AQ;
  const bf16_t* qrow = qA + ((size_t)b * NTOK + ib + m) * QKDIM;
  AQ.h[0] = *(const v8bf*)(qrow + 8 * g);
  AQ.h[1] = *(const v8bf*)(qrow + 16 + 8 * g);

  v8f acc[16];
#pragma unroll
  for (int t = 0; t < 16; ++t) acc[t] = v8f_zero();
  float mrun[8], lrun[8];
#pragma unroll
  for (int v = 0; v < 8; ++v) { mrun[v] = -1e30f; lrun[v] = 0.0f; }

  const bf16_t* kg = kM + (size_t)b * QKDIM * NTOK;  // [32][4096]
  const bf16_t* vg = vT + (size_t)b * NTOK * CDIM;   // [4096][256]

  // Prologue: stage tile 0 into buffer 0 (K: 32x32, V: 32x256, 4B units).
  if (wid == 0) {
    tdm_load_2d(kg, &Kbuf[0][0], /*tile_d0=*/16, /*tile_d1=*/32,
                /*tensor_d0=*/2048, /*tensor_d1=*/32, /*stride=*/2048);
    tdm_load_2d(vg, &Vbuf[0][0], 128, 32, 128, NTOK, 128);
    __builtin_amdgcn_s_wait_tensorcnt(0);
  }
  __syncthreads();

  for (int it = 0; it < NTOK / 32; ++it) {
    const int cur = it & 1, nxt = cur ^ 1;
    const int j0  = it * 32;

    // Kick off DMA for the next tile while this one is computed.
    if (wid == 0 && it + 1 < NTOK / 32) {
      tdm_load_2d(kg + (j0 + 32), &Kbuf[nxt][0], 16, 32, 2048, 32, 2048);
      tdm_load_2d(vg + (size_t)(j0 + 32) * CDIM, &Vbuf[nxt][0],
                  128, 32, 128, NTOK, 128);
    }

    // ---- scores: S[16 x 32] = Q @ K from LDS, two 16-col WMMAs ----------
    const bf16_t* kr = &Kbuf[cur][m];                // lane%16 = key col j, elems = qk-dim d
    BF16x16 BK0, BK1;
#pragma unroll
    for (int e = 0; e < 8; ++e) {
      BK0.v[e]     = kr[(8 * g + e) * 32];
      BK0.v[e + 8] = kr[(16 + 8 * g + e) * 32];
      BK1.v[e]     = kr[(8 * g + e) * 32 + 16];
      BK1.v[e + 8] = kr[(16 + 8 * g + e) * 32 + 16];
    }
    v8f s0 = __builtin_amdgcn_wmma_f32_16x16x32_bf16(false, AQ.v, false, BK0.v,
                                                     (short)0, v8f_zero(), false, false);
    v8f s1 = __builtin_amdgcn_wmma_f32_16x16x32_bf16(false, AQ.v, false, BK1.v,
                                                     (short)0, v8f_zero(), false, false);

    // ---- online softmax (row = v + 8*g, cols across 16-lane group) ------
    float scale[8];
#pragma unroll
    for (int v = 0; v < 8; ++v) {
      float t = fmaxf(s0[v], s1[v]);
      t = fmaxf(t, __shfl_xor(t, 1, 32));
      t = fmaxf(t, __shfl_xor(t, 2, 32));
      t = fmaxf(t, __shfl_xor(t, 4, 32));
      t = fmaxf(t, __shfl_xor(t, 8, 32));
      const float mnew = fmaxf(mrun[v], t);
      scale[v] = __expf(mrun[v] - mnew);
      const float p0 = __expf(s0[v] - mnew);
      const float p1 = __expf(s1[v] - mnew);
      s0[v] = p0; s1[v] = p1;
      float rs = p0 + p1;
      rs += __shfl_xor(rs, 1, 32);
      rs += __shfl_xor(rs, 2, 32);
      rs += __shfl_xor(rs, 4, 32);
      rs += __shfl_xor(rs, 8, 32);
      lrun[v] = lrun[v] * scale[v] + rs;
      mrun[v] = mnew;
    }
#pragma unroll
    for (int t = 0; t < 16; ++t)
#pragma unroll
      for (int v = 0; v < 8; ++v) acc[t][v] *= scale[v];

    // ---- C-layout P -> LDS -> A-layout bf16 (per-wave region) -----------
#pragma unroll
    for (int v = 0; v < 8; ++v) {
      Plds[wid][(v + 8 * g) * 32 + m]      = (bf16_t)s0[v];
      Plds[wid][(v + 8 * g) * 32 + 16 + m] = (bf16_t)s1[v];
    }
    __syncthreads();                                 // order P store -> P load
    BF16x16 AP;
    AP.h[0] = *(const v8bf*)(&Plds[wid][m * 32 + 8 * g]);
    AP.h[1] = *(const v8bf*)(&Plds[wid][m * 32 + 16 + 8 * g]);

    // ---- O[16 x 256] += P @ V^T from LDS (16 WMMAs over channel tiles) --
    const bf16_t* vrow = &Vbuf[cur][m];              // lane%16 = channel col, elems = key j
#pragma unroll
    for (int t = 0; t < 16; ++t) {
      BF16x16 BV;
#pragma unroll
      for (int e = 0; e < 8; ++e) {
        BV.v[e]     = vrow[(8 * g + e) * 256 + t * 16];
        BV.v[e + 8] = vrow[(16 + 8 * g + e) * 256 + t * 16];
      }
      acc[t] = __builtin_amdgcn_wmma_f32_16x16x32_bf16(false, AP.v, false, BV.v,
                                                       (short)0, acc[t], false, false);
    }

    // Drain the DMA for the next tile, then make it visible to all waves.
    if (wid == 0) __builtin_amdgcn_s_wait_tensorcnt(0);
    __syncthreads();
  }

  // ---- finalize: out = gamma * (O / l) + x ------------------------------
  const float gam = gamma[0];
  float inv[8];
#pragma unroll
  for (int v = 0; v < 8; ++v) inv[v] = 1.0f / lrun[v];
#pragma unroll
  for (int t = 0; t < 16; ++t) {
    const int c = t * 16 + m;
    const size_t base = ((size_t)b * CDIM + c) * NTOK + (size_t)(ib + 8 * g);
    const float4 x0 = *(const float4*)(xin + base);
    const float4 x1 = *(const float4*)(xin + base + 4);
    float4 r0, r1;
    r0.x = gam * acc[t][0] * inv[0] + x0.x;
    r0.y = gam * acc[t][1] * inv[1] + x0.y;
    r0.z = gam * acc[t][2] * inv[2] + x0.z;
    r0.w = gam * acc[t][3] * inv[3] + x0.w;
    r1.x = gam * acc[t][4] * inv[4] + x1.x;
    r1.y = gam * acc[t][5] * inv[5] + x1.y;
    r1.z = gam * acc[t][6] * inv[6] + x1.z;
    r1.w = gam * acc[t][7] * inv[7] + x1.w;
    *(float4*)(outp + base)     = r0;
    *(float4*)(outp + base + 4) = r1;
  }
}

// ---------------------------------------------------------------------------
// Launcher
// ---------------------------------------------------------------------------
extern "C" void kernel_launch(void* const* d_in, const int* in_sizes, int n_in,
                              void* d_out, int out_size, void* d_ws, size_t ws_size,
                              hipStream_t stream) {
  (void)in_sizes; (void)n_in; (void)out_size; (void)ws_size;
  const float* x     = (const float*)d_in[0];   // [4,2,256,64,64] -> [8][256][4096]
  const float* Wq    = (const float*)d_in[1];   // [32][256]
  const float* bq    = (const float*)d_in[2];   // [32]
  const float* Wk    = (const float*)d_in[3];   // [32][256]
  const float* bk    = (const float*)d_in[4];   // [32]
  const float* Wv    = (const float*)d_in[5];   // [256][256]
  const float* bv    = (const float*)d_in[6];   // [256]
  const float* gamma = (const float*)d_in[7];   // [1]
  float* out = (float*)d_out;

  // Workspace partition (bytes), total ~37.9 MB.
  char* ws = (char*)d_ws;
  bf16_t* xT  = (bf16_t*)(ws);                        // 8*4096*256*2 = 16,777,216
  bf16_t* vT  = (bf16_t*)(ws + 16777216);             // 16,777,216
  bf16_t* qA  = (bf16_t*)(ws + 33554432);             // 8*4096*32*2 = 2,097,152
  bf16_t* kM  = (bf16_t*)(ws + 35651584);             // 2,097,152
  bf16_t* WqT = (bf16_t*)(ws + 37748736);             // 16,384
  bf16_t* WvT = (bf16_t*)(ws + 37765120);             // 131,072

  // 0) x -> xT (bf16, token-major)
  k_transpose_cvt<<<dim3(NTOK / 32, CDIM / 32, FBATCH), dim3(32, 8), 0, stream>>>(x, xT);

  // 1) weight transposes (bf16)
  k_transpose_w<<<(QKDIM * CDIM + 255) / 256, 256, 0, stream>>>(Wq, WqT, QKDIM, CDIM);
  k_transpose_w<<<(CDIM * CDIM + 255) / 256, 256, 0, stream>>>(Wv, WvT, CDIM, CDIM);

  // 2) q = xT @ WqT  -> [FB][N][32];   vT = xT @ WvT -> [FB][N][256]
  k_proj_rowmajor<<<dim3(NTOK / 16, QKDIM / 16, FBATCH), 32, 0, stream>>>(xT, WqT, bq, qA, QKDIM);
  k_proj_rowmajor<<<dim3(NTOK / 16, CDIM / 16, FBATCH), 32, 0, stream>>>(xT, WvT, bv, vT, CDIM);

  // 3) k = Wk @ x -> [FB][32][N]
  k_proj_k<<<dim3(NTOK / 16, QKDIM / 16, FBATCH), 32, 0, stream>>>(x, Wk, bk, kM);

  // 4) fused flash attention + residual (4 waves / block, TDM-staged K/V)
  k_flash_attn<<<dim3(NTOK / 64, FBATCH), 128, 0, stream>>>(qA, kM, vT, x, gamma, out);
}
